// TrackletPredictionModel_67774583931429
// MI455X (gfx1250) — matrix operand order repaired and checked by reading, not compiled, not run on hardware
//
#include <hip/hip_runtime.h>
#include <hip/hip_bf16.h>

// Problem constants (fixed by the reference)
#define Bz 32
#define Tt 8192
#define Ii 16
#define Hh 128
#define Oo 3
#define Pp 1024
#define G4 512      // 4*H gate columns
#define KC 160      // K = H(128) + I(16) + 16 zero pad -> 5 k-tiles of 32

typedef __attribute__((ext_vector_type(16))) __bf16       v16bf;
typedef __attribute__((ext_vector_type(8)))  float        v8f;
typedef __attribute__((ext_vector_type(4)))  unsigned int v4u;

union Fr { v16bf v; v4u u[2]; };

// CDNA5 hardware transcendental tanh (TRANS32 group). Trailing v_nop covers
// the ISA rule "1 independent op or V_NOP after a TRANS op before the output
// is used" since the compiler cannot model inline-asm latency.
__device__ __forceinline__ float tanh_hw(float x) {
    float r;
    asm volatile("v_tanh_f32 %0, %1\n\tv_nop" : "=v"(r) : "v"(x));
    return r;
}

// Branch-free sigmoid via hardware tanh: sigmoid(x) = 0.5*tanh(x/2) + 0.5
__device__ __forceinline__ float sig_hw(float x) {
    return __builtin_fmaf(0.5f, tanh_hw(0.5f * x), 0.5f);
}

// ---------------------------------------------------------------------------
// Kernel 1: zero the segment accumulators (sums[3] + count per (b, photo))
// ---------------------------------------------------------------------------
__global__ void zero_kernel(float* __restrict__ acc) {
    int i = blockIdx.x * blockDim.x + threadIdx.x;
    if (i < Bz * Pp * 4) acc[i] = 0.0f;
}

// ---------------------------------------------------------------------------
// Kernel 2: fused LSTM recurrence with WMMA.
//   grid = 2 workgroups (16 batch rows each), 256 threads = 8 waves.
//   Wave w owns gate tiles {w, 8+w, 16+w, 24+w} = matched (i,f,g,o) columns
//   for hidden units [16w, 16w+16) -> cell state stays in registers.
//   h is double-buffered in LDS in the ISA 16-bit A-operand striping so each
//   A fragment is two ds_load_b128; Wcat B fragments are register-resident.
// ---------------------------------------------------------------------------
__launch_bounds__(256)
__global__ void lstm_kernel(const float* __restrict__ x,
                            const float* __restrict__ W_ih,
                            const float* __restrict__ W_hh,
                            const float* __restrict__ b_ih,
                            const float* __restrict__ b_hh,
                            __bf16* __restrict__ h_buf) {
    extern __shared__ __bf16 smem[];
    __bf16* Wc    = smem;               // [512][160] bf16 = 160 KB
    __bf16* abuf0 = smem + G4 * KC;     // [16][160]  (h | x | pad), buffer 0
    __bf16* abuf1 = abuf0 + 16 * KC;    // buffer 1

    const int tid  = threadIdx.x;
    const int lane = tid & 31;
    const int w    = tid >> 5;          // wave 0..7
    const int n    = lane & 15;
    const int hi   = lane >> 4;         // 0: lanes 0-15, 1: lanes 16-31
    const int b0   = blockIdx.x * 16;   // batch base

    // ---- Build Wcat = [W_hh | W_ih | 0] in LDS as bf16, row-major [512][160]
    for (int r = tid; r < G4; r += 256) {
        const float* wh = W_hh + r * Hh;
        const float* wi = W_ih + r * Ii;
        __bf16* dst = Wc + r * KC;
        for (int k = 0; k < Hh; ++k)       dst[k]      = (__bf16)wh[k];
        for (int k = 0; k < Ii; ++k)       dst[Hh + k] = (__bf16)wi[k];
        for (int k = Hh + Ii; k < KC; ++k) dst[k]      = (__bf16)0.0f;
    }
    // ---- Zero both a-buffers (avoids NaN bit-patterns: WMMA 0*NaN = NaN)
    for (int i = tid; i < 2 * 16 * KC; i += 256) abuf0[i] = (__bf16)0.0f;
    __syncthreads();

    // ---- Load register-resident B fragments + fused bias, per the ISA
    //      16-bit B 32x16 layout: lanes 0-15 hold K 0..7,16..23, lanes 16-31 K 8..15,24..31
    Fr    Bf[4][5];
    float biasv[4];
    for (int g = 0; g < 4; ++g) {
        const int ntile = 8 * g + w;          // gate g, wave's tile
        const int row   = ntile * 16 + n;     // gate column 0..511
        biasv[g] = b_ih[row] + b_hh[row];
        for (int kt = 0; kt < 5; ++kt) {
            const v4u* p = (const v4u*)(Wc + row * KC + kt * 32 + hi * 8);
            Bf[g][kt].u[0] = p[0];
            Bf[g][kt].u[1] = p[2];
        }
    }

    // ---- Stage x_0 into buffer 0 (h region already zero = h_0)
    {
        const int m = tid >> 4, ch = tid & 15;
        float xv = x[(((size_t)(b0 + m)) * Tt + 0) * Ii + ch];
        abuf0[m * KC + Hh + ch] = (__bf16)xv;
    }
    __syncthreads();

    v8f creg;
    for (int j = 0; j < 8; ++j) creg[j] = 0.0f;

    for (int t = 0; t < Tt; ++t) {
        __bf16* cur = (t & 1) ? abuf1 : abuf0;
        __bf16* nxt = (t & 1) ? abuf0 : abuf1;

        // Stage x_{t+1} into the other buffer; prefetch x_{t+2}.
        {
            const int m = tid >> 4, ch = tid & 15;
            const int tn = (t + 1 < Tt) ? (t + 1) : t;
            float xv = x[(((size_t)(b0 + m)) * Tt + tn) * Ii + ch];
            nxt[m * KC + Hh + ch] = (__bf16)xv;
            const int tp = (t + 2 < Tt) ? (t + 2) : t;
            __builtin_prefetch(&x[(((size_t)(b0 + m)) * Tt + tp) * Ii + ch], 0, 1);
        }

        // A fragments: ISA 16-bit A 16x32 striping -> 2x ds_load_b128 each.
        Fr Af[5];
        {
            const int koff = hi ? 8 : 0;      // lanes>=16 hold K 8..15 / 24..31
            for (int kt = 0; kt < 5; ++kt) {
                const __bf16* base = cur + n * KC + kt * 32 + koff;
                Af[kt].u[0] = *(const v4u*)(base);
                Af[kt].u[1] = *(const v4u*)(base + 16);
            }
        }

        // 20 WMMAs: 4 gate tiles x 5 k-tiles, C seeded with fused bias.
        v8f D[4];
        for (int g = 0; g < 4; ++g) {
            v8f d;
            for (int j = 0; j < 8; ++j) d[j] = biasv[g];
            for (int kt = 0; kt < 5; ++kt)
                d = __builtin_amdgcn_wmma_f32_16x16x32_bf16(
                        false, Af[kt].v, false, Bf[g][kt].v,
                        (short)0, d, false, false);
            D[g] = d;
        }

        // Gate nonlinearities + cell update: branch-free, all on the CDNA5
        // hardware v_tanh_f32 transcendental (EXEC stays all-1s).
        __bf16 hb[8];
        for (int j = 0; j < 8; ++j) {
            float ig = sig_hw(D[0][j]);
            float fg = sig_hw(D[1][j]);
            float gg = tanh_hw(D[2][j]);
            float og = sig_hw(D[3][j]);
            float c  = __builtin_fmaf(fg, creg[j], ig * gg);
            creg[j]  = c;
            hb[j]    = (__bf16)(og * tanh_hw(c));
        }

        // Scatter h_t into next a-buffer (row-major [m][hid]) + spill to HBM.
        {
            const int hid = w * 16 + n;
            for (int j = 0; j < 8; ++j) {
                const int m = hi ? (j + 8) : j;   // C/D layout M mapping
                nxt[m * KC + hid] = hb[j];
                h_buf[(((size_t)(b0 + m)) * Tt + t) * Hh + hid] = hb[j];
            }
        }
        __syncthreads();   // one barrier per step (double-buffered)
    }
}

// ---------------------------------------------------------------------------
// Kernel 3: FC head (128 -> 3) + segment accumulation via f32 atomics.
//   One thread per (b, t). Reads bf16 h (64 MB), W_fc cached in LDS.
// ---------------------------------------------------------------------------
__launch_bounds__(256)
__global__ void head_kernel(const float* __restrict__ x,
                            const __bf16* __restrict__ h_buf,
                            const float* __restrict__ W_fc,
                            const float* __restrict__ b_fc,
                            float* __restrict__ acc) {
    __shared__ float wfs[Oo * Hh];
    for (int i = threadIdx.x; i < Oo * Hh; i += 256) wfs[i] = W_fc[i];
    __syncthreads();

    const size_t g = (size_t)blockIdx.x * 256 + threadIdx.x;   // = b*T + t
    const int b = (int)(g >> 13);

    float a0 = b_fc[0], a1 = b_fc[1], a2 = b_fc[2];
    const unsigned int* hp = (const unsigned int*)(h_buf + g * Hh);
    for (int k2 = 0; k2 < Hh / 2; ++k2) {
        const unsigned int u = hp[k2];
        const float h0 = __uint_as_float(u << 16);
        const float h1 = __uint_as_float(u & 0xffff0000u);
        const int k = 2 * k2;
        a0 += h0 * wfs[k]          + h1 * wfs[k + 1];
        a1 += h0 * wfs[Hh + k]     + h1 * wfs[Hh + k + 1];
        a2 += h0 * wfs[2 * Hh + k] + h1 * wfs[2 * Hh + k + 1];
    }

    const int id = (int)x[g * Ii + 2];   // photo id lives in feature 2
    if (id >= 0 && id < Pp) {
        float* a = acc + ((size_t)b * Pp + id) * 4;
        atomicAdd(a + 0, a0);
        atomicAdd(a + 1, a1);
        atomicAdd(a + 2, a2);
        atomicAdd(a + 3, 1.0f);
    }
}

// ---------------------------------------------------------------------------
// Kernel 4: mean with count==0 -> 1 semantics.
// ---------------------------------------------------------------------------
__global__ void finalize_kernel(const float* __restrict__ acc,
                                float* __restrict__ out) {
    const int i = blockIdx.x * blockDim.x + threadIdx.x;
    if (i >= Bz * Pp) return;
    const float* a = acc + (size_t)i * 4;
    float cnt = a[3];
    cnt = (cnt == 0.0f) ? 1.0f : cnt;
    out[i * 3 + 0] = a[0] / cnt;
    out[i * 3 + 1] = a[1] / cnt;
    out[i * 3 + 2] = a[2] / cnt;
}

// ---------------------------------------------------------------------------
extern "C" void kernel_launch(void* const* d_in, const int* in_sizes, int n_in,
                              void* d_out, int out_size, void* d_ws, size_t ws_size,
                              hipStream_t stream) {
    (void)in_sizes; (void)n_in; (void)out_size; (void)ws_size;
    const float* x   = (const float*)d_in[0];
    const float* Wih = (const float*)d_in[1];
    const float* Whh = (const float*)d_in[2];
    const float* bih = (const float*)d_in[3];
    const float* bhh = (const float*)d_in[4];
    const float* Wfc = (const float*)d_in[5];
    const float* bfc = (const float*)d_in[6];

    float*  acc   = (float*)d_ws;                                     // 512 KB
    __bf16* h_buf = (__bf16*)((char*)d_ws +
                              (size_t)Bz * Pp * 4 * sizeof(float));   // 64 MB

    hipLaunchKernelGGL(zero_kernel, dim3((Bz * Pp * 4 + 255) / 256), dim3(256),
                       0, stream, acc);

    const size_t shmem = (size_t)(G4 * KC + 2 * 16 * KC) * sizeof(__bf16); // 170 KB
    hipLaunchKernelGGL(lstm_kernel, dim3(2), dim3(256), shmem, stream,
                       x, Wih, Whh, bih, bhh, h_buf);

    hipLaunchKernelGGL(head_kernel, dim3((Bz * Tt) / 256), dim3(256), 0, stream,
                       x, h_buf, Wfc, bfc, acc);

    hipLaunchKernelGGL(finalize_kernel, dim3((Bz * Pp + 255) / 256), dim3(256),
                       0, stream, acc, (float*)d_out);
}
